// ExpertParallelWrapper_42606075577078
// MI455X (gfx1250) — compile-verified, matrix operand order checked by reading, not run on hardware
//
#include <hip/hip_runtime.h>
#include <hip/hip_bf16.h>
#include <math.h>

typedef __attribute__((ext_vector_type(16))) __bf16        v16bf;
typedef __attribute__((ext_vector_type(8)))  float         v8f;
typedef __attribute__((ext_vector_type(8)))  unsigned int  v8u;

constexpr int kBatch = 2, kS = 2048, kH = 1024, kF = 2816, kE = 8, kTK = 2;
constexpr int kN = kBatch * kS;  // 4096 tokens
constexpr int kA = kN * kTK;     // 8192 (token, expert) assignments

constexpr int kLdsStride = 36;   // dwords per LDS tile row (144B: 16B-aligned rows)

// ---------- helpers ----------
__device__ __forceinline__ unsigned short f2bf(float f) {
  unsigned int u = __builtin_bit_cast(unsigned int, f);
  u += 0x7fffu + ((u >> 16) & 1u);          // round-to-nearest-even
  return (unsigned short)(u >> 16);
}
__device__ __forceinline__ unsigned pack2bf(float a, float b) {
  return (unsigned)f2bf(a) | ((unsigned)f2bf(b) << 16);
}

// Fragment-ready LDS permutation. A tile row holds 32 dwords (64 bf16 K values)
// in slot order per 32-K block: K-pairs [0..3][8..11][4..7][12..15]. Then a
// WMMA fragment (ISA 16-bit layout: lanes 0-15 need K kb+0..7 and kb+16..23,
// kb=0; lanes 16-31 kb=8) is one contiguous 32B chunk:
//   lane<16 -> block dwords 0..7, lane>=16 -> block dwords 8..15.
__device__ __forceinline__ int perm_slot(int pairIdx /*0..31*/) {
  int blk = pairIdx >> 4;              // 32-K block (0: K 0-31, 1: K 32-63)
  int q = pairIdx & 15;
  int m = q >> 2;                      // 0,1,2,3
  int mp = ((m & 1) << 1) | (m >> 1);  // -> 0,2,1,3
  return blk * 16 + (mp << 2) + (q & 3);
}

__device__ __forceinline__ v16bf frag_ld(const unsigned* T, int row, int ks, int lane) {
  const uint4* p = (const uint4*)(T + row * kLdsStride + (ks >> 1) + ((lane & 16) ? 8 : 0));
  uint4 a = p[0], b = p[1];            // two ds_load_b128, contiguous dst
  v8u u;
  u[0] = a.x; u[1] = a.y; u[2] = a.z; u[3] = a.w;
  u[4] = b.x; u[5] = b.y; u[6] = b.z; u[7] = b.w;
  return __builtin_bit_cast(v16bf, u);
}

// ---------- router ----------
__global__ __launch_bounds__(256) void moe_zero(int* counts) {
  if (threadIdx.x < kE) counts[threadIdx.x] = 0;
}

__global__ __launch_bounds__(256) void moe_router(
    const float* __restrict__ x, const float* __restrict__ gw,
    int* __restrict__ topi, float* __restrict__ topw, int* __restrict__ counts) {
  int t = blockIdx.x * 256 + threadIdx.x;
  if (t >= kN) return;
  float l[kE];
#pragma unroll
  for (int e = 0; e < kE; ++e) l[e] = 0.f;
  const float* xr = x + (size_t)t * kH;
  for (int h = 0; h < kH; ++h) {
    float xv = xr[h];
#pragma unroll
    for (int e = 0; e < kE; ++e) l[e] += xv * gw[h * kE + e];
  }
  float m = l[0];
#pragma unroll
  for (int e = 1; e < kE; ++e) m = fmaxf(m, l[e]);
  float p[kE];
#pragma unroll
  for (int e = 0; e < kE; ++e) p[e] = __expf(l[e] - m);
  int i0 = 0;
#pragma unroll
  for (int e = 1; e < kE; ++e) if (p[e] > p[i0]) i0 = e;
  int i1 = (i0 == 0) ? 1 : 0;
#pragma unroll
  for (int e = 0; e < kE; ++e) if (e != i0 && p[e] > p[i1]) i1 = e;
  float s = p[i0] + p[i1];
  topi[2 * t + 0] = i0; topw[2 * t + 0] = p[i0] / s;
  topi[2 * t + 1] = i1; topw[2 * t + 1] = p[i1] / s;
  atomicAdd(&counts[i0], 1);
  atomicAdd(&counts[i1], 1);
}

__global__ void moe_scan(const int* __restrict__ counts,
                         int* __restrict__ offs, int* __restrict__ fill) {
  if (threadIdx.x == 0) {
    int acc = 0;
    for (int e = 0; e < kE; ++e) { offs[e] = acc; fill[e] = acc; acc += counts[e]; }
    offs[kE] = acc;
  }
}

__global__ __launch_bounds__(256) void moe_fill(
    const int* __restrict__ topi, const float* __restrict__ topw,
    int* __restrict__ fill, int* __restrict__ rows,
    float* __restrict__ rw, int* __restrict__ arow) {
  int t = blockIdx.x * 256 + threadIdx.x;
  if (t >= kN) return;
#pragma unroll
  for (int k = 0; k < kTK; ++k) {
    int e = topi[2 * t + k];
    int pos = atomicAdd(&fill[e], 1);
    rows[pos] = t;
    rw[pos] = topw[2 * t + k];
    arow[2 * t + k] = pos;
  }
}

// ---------- stage A: h = silu(x*Wg) * (x*Wu), grouped per expert ----------
// Block tile 128(M) x 128(F), K-stage 64. 8 waves as 2(M) x 4(N); each wave
// owns 64x32 per output matrix -> 4x2 accum tiles for gate and up.
__global__ __launch_bounds__(256) void moe_ffn1(
    const float* __restrict__ x, const float* __restrict__ wg,
    const float* __restrict__ wu, const int* __restrict__ rows,
    const int* __restrict__ offs, unsigned short* __restrict__ hbuf) {
  __shared__ __align__(16) unsigned At[128 * kLdsStride];  // tokens  [row][k-perm]
  __shared__ __align__(16) unsigned Bgt[128 * kLdsStride]; // w_gate  [f][k-perm]
  __shared__ __align__(16) unsigned But[128 * kLdsStride]; // w_up    [f][k-perm]

  const int e = blockIdx.x;
  const int base = offs[e];
  const int cnt = offs[e + 1] - base;
  const int row0 = blockIdx.y * 128;
  if (row0 >= cnt) return;                  // block-uniform exit
  const int f0 = blockIdx.z * 128;

  const int tid = threadIdx.x, wave = tid >> 5, lane = tid & 31;
  const int rb = (wave & 1) * 64, cb = (wave >> 1) * 32;

  // Clamped token gather: rows >= cnt duplicate the last valid row; their
  // outputs are garbage but never stored. No branches in the hot loop.
  const int rA = tid >> 5;                  // A-load row base (stride 8)
  const int c2 = tid & 31;                  // A-load k-pair index
  const int slotA = perm_slot(c2);
  int srcRow[16];
#pragma unroll
  for (int j = 0; j < 16; ++j) {
    int grow = row0 + rA + 8 * j;
    srcRow[j] = rows[base + (grow < cnt ? grow : cnt - 1)];
  }
  const int fcol = tid & 127;               // B-load column within tile
  const int k2b = tid >> 7;                 // B-load k-pair base (0/1)
  int slotB[16];
#pragma unroll
  for (int j = 0; j < 16; ++j) slotB[j] = perm_slot(k2b + 2 * j);

  v8f accg[4][2] = {};
  v8f accu[4][2] = {};

  for (int kk = 0; kk < kH; kk += 64) {
#pragma unroll
    for (int j = 0; j < 16; ++j) {          // A: 128x64 gathered, fp32->bf16 packed
      int r = rA + 8 * j;
      const float2 v = *(const float2*)(x + (size_t)srcRow[j] * kH + kk + 2 * c2);
      At[r * kLdsStride + slotA] = pack2bf(v.x, v.y);
    }
#pragma unroll
    for (int j = 0; j < 16; ++j) {          // B tiles 64x128, permuted [f][k]
      int k2 = k2b + 2 * j;
      size_t i0 = ((size_t)e * kH + kk + 2 * k2) * kF + f0 + fcol;
      Bgt[fcol * kLdsStride + slotB[j]] = pack2bf(wg[i0], wg[i0 + kF]);
      But[fcol * kLdsStride + slotB[j]] = pack2bf(wu[i0], wu[i0 + kF]);
    }
    if (kk + 64 < kH) {                     // stream-ahead hint for weights
      size_t pi = ((size_t)e * kH + kk + 64 + 2 * k2b) * kF + f0 + fcol;
      __builtin_prefetch(wg + pi, 0, 1);
      __builtin_prefetch(wu + pi, 0, 1);
    }
    __syncthreads();

#pragma unroll
    for (int ks = 0; ks < 64; ks += 32) {
      v16bf af[4], bgf[2], bum[2];
#pragma unroll
      for (int mm = 0; mm < 4; ++mm)
        af[mm] = frag_ld(At, rb + mm * 16 + (lane & 15), ks, lane);
#pragma unroll
      for (int nn = 0; nn < 2; ++nn) {
        bgf[nn] = frag_ld(Bgt, cb + nn * 16 + (lane & 15), ks, lane);
        bum[nn] = frag_ld(But, cb + nn * 16 + (lane & 15), ks, lane);
      }
#pragma unroll
      for (int mm = 0; mm < 4; ++mm)
#pragma unroll
        for (int nn = 0; nn < 2; ++nn) {
          accg[mm][nn] = __builtin_amdgcn_wmma_f32_16x16x32_bf16(
              false, af[mm], false, bgf[nn], (short)0, accg[mm][nn], false, false);
          accu[mm][nn] = __builtin_amdgcn_wmma_f32_16x16x32_bf16(
              false, af[mm], false, bum[nn], (short)0, accu[mm][nn], false, false);
        }
    }
    __syncthreads();
  }

  const int lrow = (lane & 16) ? 8 : 0, lcol = lane & 15;
#pragma unroll
  for (int mm = 0; mm < 4; ++mm)
#pragma unroll
    for (int nn = 0; nn < 2; ++nn) {
      int gr0 = row0 + rb + mm * 16 + lrow;
      int gc = f0 + cb + nn * 16 + lcol;
#pragma unroll
      for (int i = 0; i < 8; ++i) {
        int grow = gr0 + i;
        if (grow < cnt) {
          float g = accg[mm][nn][i], u = accu[mm][nn][i];
          float hv = (g / (1.f + __expf(-g))) * u;   // silu(g) * u
          hbuf[(size_t)(base + grow) * kF + gc] = f2bf(hv);
        }
      }
    }
}

// ---------- stage B: tmp[row] = w_row * (h[row] @ Wd), grouped per expert ----
// Block tile 128(M) x 128(H), K-stage 64 over F. 8 waves as 2x4, wave = 64x32.
// The A tile is a raw bf16 copy -> use the CDNA5 async global->LDS data path.
__global__ __launch_bounds__(256) void moe_ffn2(
    const unsigned short* __restrict__ hbuf, const float* __restrict__ wd,
    const int* __restrict__ offs, const float* __restrict__ rw,
    float* __restrict__ tmp) {
  __shared__ __align__(16) unsigned At[128 * kLdsStride];
  __shared__ __align__(16) unsigned Bdt[128 * kLdsStride];

  const int e = blockIdx.x;
  const int base = offs[e];
  const int cnt = offs[e + 1] - base;
  const int row0 = blockIdx.y * 128;
  if (row0 >= cnt) return;
  const int h0 = blockIdx.z * 128;

  const int tid = threadIdx.x, wave = tid >> 5, lane = tid & 31;
  const int rb = (wave & 1) * 64, cb = (wave >> 1) * 32;

  const int rA = tid >> 5;
  const int c2 = tid & 31;
  const int slotA = perm_slot(c2);
  int srcIdx[16];                           // clamped h-buffer rows
  unsigned ldsA[16];                        // per-lane LDS byte dest (permuted)
#pragma unroll
  for (int j = 0; j < 16; ++j) {
    int grow = row0 + rA + 8 * j;
    srcIdx[j] = base + (grow < cnt ? grow : cnt - 1);
    ldsA[j] = (unsigned)(size_t)&At[(rA + 8 * j) * kLdsStride + slotA];
  }
  const int hc = tid & 127;
  const int k2b = tid >> 7;
  int slotB[16];
#pragma unroll
  for (int j = 0; j < 16; ++j) slotB[j] = perm_slot(k2b + 2 * j);

  v8f acc[4][2] = {};

  for (int kk = 0; kk < kF; kk += 64) {
#pragma unroll
    for (int j = 0; j < 16; ++j) {          // A: async bf16 dword copy to LDS
      unsigned long long ga =
          (unsigned long long)(size_t)(hbuf + (size_t)srcIdx[j] * kF + kk + 2 * c2);
      asm volatile("global_load_async_to_lds_b32 %0, %1, off"
                   :: "v"(ldsA[j]), "v"(ga) : "memory");
    }
#pragma unroll
    for (int j = 0; j < 16; ++j) {          // Wd tile 64x128, permuted [h][k]
      int k2 = k2b + 2 * j;
      size_t i0 = ((size_t)e * kF + kk + 2 * k2) * kH + h0 + hc;
      Bdt[hc * kLdsStride + slotB[j]] = pack2bf(wd[i0], wd[i0 + kH]);
    }
    if (kk + 64 < kF) {
      size_t pi = ((size_t)e * kF + kk + 64 + 2 * k2b) * kH + h0 + hc;
      __builtin_prefetch(wd + pi, 0, 1);
    }
    asm volatile("s_wait_asynccnt 0" ::: "memory");
    __syncthreads();

#pragma unroll
    for (int ks = 0; ks < 64; ks += 32) {
      v16bf af[4], bf[2];
#pragma unroll
      for (int mm = 0; mm < 4; ++mm)
        af[mm] = frag_ld(At, rb + mm * 16 + (lane & 15), ks, lane);
#pragma unroll
      for (int nn = 0; nn < 2; ++nn)
        bf[nn] = frag_ld(Bdt, cb + nn * 16 + (lane & 15), ks, lane);
#pragma unroll
      for (int mm = 0; mm < 4; ++mm)
#pragma unroll
        for (int nn = 0; nn < 2; ++nn)
          acc[mm][nn] = __builtin_amdgcn_wmma_f32_16x16x32_bf16(
              false, af[mm], false, bf[nn], (short)0, acc[mm][nn], false, false);
    }
    __syncthreads();
  }

  const int lrow = (lane & 16) ? 8 : 0, lcol = lane & 15;
#pragma unroll
  for (int mm = 0; mm < 4; ++mm)
#pragma unroll
    for (int nn = 0; nn < 2; ++nn) {
      int gr0 = row0 + rb + mm * 16 + lrow;
      int gc = h0 + cb + nn * 16 + lcol;
#pragma unroll
      for (int i = 0; i < 8; ++i) {
        int grow = gr0 + i;
        if (grow < cnt) {
          float w = rw[base + grow];
          tmp[(size_t)(base + grow) * kH + gc] = w * acc[mm][nn][i];
        }
      }
    }
}

// ---------- deterministic 2-way combine ----------
__global__ __launch_bounds__(256) void moe_combine(
    const float* __restrict__ tmp, const int* __restrict__ arow,
    float* __restrict__ out) {
  int idx = blockIdx.x * 256 + threadIdx.x;
  if (idx >= kN * kH) return;
  int t = idx >> 10;            // / kH
  int c = idx & (kH - 1);
  int a0 = arow[2 * t + 0], a1 = arow[2 * t + 1];
  out[idx] = tmp[(size_t)a0 * kH + c] + tmp[(size_t)a1 * kH + c];
}

// ---------- launcher ----------
extern "C" void kernel_launch(void* const* d_in, const int* in_sizes, int n_in,
                              void* d_out, int out_size, void* d_ws, size_t ws_size,
                              hipStream_t stream) {
  const float* x  = (const float*)d_in[0];   // [B,S,H]
  const float* gw = (const float*)d_in[1];   // [H,E]
  const float* wg = (const float*)d_in[2];   // [E,H,F]
  const float* wu = (const float*)d_in[3];   // [E,H,F]
  const float* wd = (const float*)d_in[4];   // [E,F,H]
  (void)in_sizes; (void)n_in; (void)out_size; (void)ws_size;

  char* ws = (char*)d_ws;
  int*   counts = (int*)(ws + 0);
  int*   fill   = (int*)(ws + 256);
  int*   offs   = (int*)(ws + 512);
  int*   topi   = (int*)(ws + 1024);
  float* topw   = (float*)(ws + 1024 + 4 * (size_t)kA);
  int*   rows   = (int*)(ws + 1024 + 8 * (size_t)kA);
  float* rw     = (float*)(ws + 1024 + 12 * (size_t)kA);
  int*   arow   = (int*)(ws + 1024 + 16 * (size_t)kA);
  unsigned short* hbuf = (unsigned short*)(ws + 1024 + 20 * (size_t)kA);
  float* tmp    = (float*)(ws + 1024 + 20 * (size_t)kA + 2 * (size_t)kA * kF);
  float* out    = (float*)d_out;

  moe_zero<<<1, 256, 0, stream>>>(counts);
  moe_router<<<kN / 256, 256, 0, stream>>>(x, gw, topi, topw, counts);
  moe_scan<<<1, 1, 0, stream>>>(counts, offs, fill);
  moe_fill<<<kN / 256, 256, 0, stream>>>(topi, topw, fill, rows, rw, arow);

  dim3 g1(kE, kN / 128, kF / 128);           // 8 x 32 x 22
  moe_ffn1<<<g1, 256, 0, stream>>>(x, wg, wu, rows, offs, hbuf);

  dim3 g2(kE, kN / 128, kH / 128);           // 8 x 32 x 8
  moe_ffn2<<<g2, 256, 0, stream>>>(hbuf, wd, offs, rw, tmp);

  moe_combine<<<(kN * kH) / 256, 256, 0, stream>>>(tmp, arow, out);
}
